// Ferminet_85968065397338
// MI455X (gfx1250) — matrix operand order repaired and checked by reading, not compiled, not run on hardware
//
#include <hip/hip_runtime.h>
#include <hip/hip_bf16.h>

typedef __attribute__((ext_vector_type(16))) __bf16 v16bf;
typedef __attribute__((ext_vector_type(8)))  __bf16 v8bf;
typedef __attribute__((ext_vector_type(8)))  float  v8f;

#define NE   16   // electrons
#define S0E  8    // spin-up electrons
#define NM   4    // nucleons
#define ND   16   // determinants
#define NO1  256  // one-stream width
#define NO2  32   // two-stream width
#define FMAX 832  // max feature width (3*256 + 2*32)

// ---- packed-weight workspace: FRAGMENT-ORDER bf16 ------------------------
// one-stream:  [ct][ks][lane][16]  (lane fragment = 32B contiguous, coalesced)
// two-stream:  [ct][lane][16]
// env:         [tile][ks][lane][16], pre-swizzled to A-register element order
#define OFF_V0 0                          // L0: 16ct x 2ks x 512   = 16384
#define OFF_V  16384                      // 3 layers x 16ct x 26ks x 512
#define VL_SZ  (16*26*512)                // 212992 per layer
#define OFF_W  (OFF_V + 3*VL_SZ)          // 4 layers x 2ct x 512   = 4096
#define OFF_E  (OFF_W + 4*1024)           // 8t x 8ks x 512         = 32768
#define WS_ELEMS (OFF_E + 8*8*512)        // 692224
#define WS_BYTES ((size_t)WS_ELEMS * 2)   // 1384448

// ---------------------------------------------------------------------------
__device__ __forceinline__ v8f wmma_bf16(v16bf a, v16bf b, v8f c) {
  return __builtin_amdgcn_wmma_f32_16x16x32_bf16(
      false, a, false, b, (short)0, c, false, false);
}

__device__ __forceinline__ float fast_tanh(float x) {   // branchless
  float t = __expf(2.f * x);
  return 1.f - 2.f * __builtin_amdgcn_rcpf(t + 1.f);
}

// 16 contiguous bf16 -> fragment register (two aligned 16B loads).
__device__ __forceinline__ v16bf frag16(const __bf16* p) {
  v8bf lo = *(const v8bf*)p;
  v8bf hi = *(const v8bf*)(p + 8);
  v16bf r;
#pragma unroll
  for (int e = 0; e < 8; ++e) { r[e] = lo[e]; r[e + 8] = hi[e]; }
  return r;
}

// A fragment from LDS bf16 row: element e -> K = e + 2*hi8.
__device__ __forceinline__ v16bf fragA_bf16(const __bf16* rowkb, int hi8) {
  v8bf lo = *(const v8bf*)(rowkb + 2 * hi8);
  v8bf hi = *(const v8bf*)(rowkb + 8 + 2 * hi8);
  v16bf r;
#pragma unroll
  for (int e = 0; e < 8; ++e) { r[e] = lo[e]; r[e + 8] = hi[e]; }
  return r;
}

// ---- f32 loaders (fallbacks when ws_size is too small; fragA_f32 also loads two-stream A) ----
__device__ __forceinline__ v16bf fragA_f32(const float* rowkb, int hi8) {
  const float4 a = *(const float4*)(rowkb + 2 * hi8);
  const float4 b = *(const float4*)(rowkb + 2 * hi8 + 4);
  const float4 c = *(const float4*)(rowkb + 2 * hi8 + 8);
  const float4 d = *(const float4*)(rowkb + 2 * hi8 + 12);
  v16bf r;
  r[0]=(__bf16)a.x;  r[1]=(__bf16)a.y;  r[2]=(__bf16)a.z;  r[3]=(__bf16)a.w;
  r[4]=(__bf16)b.x;  r[5]=(__bf16)b.y;  r[6]=(__bf16)b.z;  r[7]=(__bf16)b.w;
  r[8]=(__bf16)c.x;  r[9]=(__bf16)c.y;  r[10]=(__bf16)c.z; r[11]=(__bf16)c.w;
  r[12]=(__bf16)d.x; r[13]=(__bf16)d.y; r[14]=(__bf16)d.z; r[15]=(__bf16)d.w;
  return r;
}
__device__ __forceinline__ v16bf fragB_f32(const float* base) {
  const float4* q = (const float4*)base;
  float4 a = q[0], b = q[1], c = q[2], d = q[3];
  v16bf r;
  r[0]=(__bf16)a.x;  r[1]=(__bf16)a.y;  r[2]=(__bf16)a.z;  r[3]=(__bf16)a.w;
  r[4]=(__bf16)b.x;  r[5]=(__bf16)b.y;  r[6]=(__bf16)b.z;  r[7]=(__bf16)b.w;
  r[8]=(__bf16)c.x;  r[9]=(__bf16)c.y;  r[10]=(__bf16)c.z; r[11]=(__bf16)c.w;
  r[12]=(__bf16)d.x; r[13]=(__bf16)d.y; r[14]=(__bf16)d.z; r[15]=(__bf16)d.w;
  return r;
}
template<int LIMIT>
__device__ __forceinline__ v16bf fragB_clamp(const float* row, int kb, int hib16) {
  v16bf r;
#pragma unroll
  for (int e = 0; e < 16; ++e) {
    int k = kb + hib16 + e;
    int kc = (k < LIMIT) ? k : (LIMIT - 1);
    float v = row[kc];
    r[e] = (__bf16)((k < LIMIT) ? v : 0.f);
  }
  return r;
}

// ---------------------------------------------------------------------------
// Weight pack kernel: f32 -> bf16 fragment-order, run once per launch.
// ---------------------------------------------------------------------------
extern "C" __global__ __launch_bounds__(256)
void pack_weights(const float* __restrict__ v0_w, const float* __restrict__ v_w,
                  const float* __restrict__ w0_w, const float* __restrict__ w_w,
                  const float* __restrict__ env_w, __bf16* __restrict__ ws) {
  int idx = blockIdx.x * 256 + threadIdx.x;
  if (idx >= WS_ELEMS) return;
  float val;
  if (idx < OFF_V) {                       // layer-0 one-stream: 16ct x 2ks
    int j = idx;
    int e = j & 15, lane = (j >> 4) & 31, t = j >> 9;
    int ks = t & 1, ct = t >> 1;
    int o = 16 * ct + (lane & 15);
    int k = ks * 32 + ((lane & 16) ? 16 : 0) + e;
    val = (k < 56) ? v0_w[o * 56 + k] : 0.f;
  } else if (idx < OFF_W) {                // layers 1-3 one-stream: 16ct x 26ks
    int j2 = idx - OFF_V;
    int layer = j2 / VL_SZ;                // 0..2
    int j = j2 - layer * VL_SZ;
    int e = j & 15, lane = (j >> 4) & 31, t = j >> 9;  // t = ct*26+ks
    int ks = t % 26, ct = t / 26;
    int o = 16 * ct + (lane & 15);
    int k = ks * 32 + ((lane & 16) ? 16 : 0) + e;
    val = v_w[((size_t)layer * NO1 + o) * FMAX + k];
  } else if (idx < OFF_E) {                // two-stream: 4 layers x 2ct
    int j2 = idx - OFF_W;
    int layer = j2 >> 10;
    int j = j2 & 1023;
    int e = j & 15, lane = (j >> 4) & 31, ct = j >> 9;
    int o = 16 * ct + (lane & 15);
    int k = ((lane & 16) ? 16 : 0) + e;
    if (layer == 0) val = (k < 4) ? w0_w[o * 4 + k] : 0.f;
    else            val = w_w[((layer - 1) * NO2 + o) * NO2 + k];
  } else {                                 // env: 8 tiles x 8 ks, A-reg order
    int j = idx - OFF_E;
    int e = j & 15, lane = (j >> 4) & 31, ks = (j >> 9) & 7, t = j >> 12;
    int r = 16 * t + (lane & 15);          // r = det*8 + orbital
    int kd = r >> 3, io = r & 7;
    int k = ks * 32 + ((lane & 16) ? 16 : 0) + e;
    val = env_w[((size_t)kd * NE + io) * NO1 + k];
  }
  ws[idx] = (__bf16)val;
}

// ---------------------------------------------------------------------------
// One message-passing layer.
// ---------------------------------------------------------------------------
template<bool FIRST, bool PACKED>
__device__ __forceinline__ void do_layer(
    bool resid,
    const __bf16* __restrict__ pV, const __bf16* __restrict__ pW,  // packed
    const float*  __restrict__ Wv, const float*  __restrict__ Ww,  // fallback
    const float*  __restrict__ bv, const float*  __restrict__ bw,
    float* sOne, float* sTwo, float* sG1u, float* sG1d,
    float* sG2u, float* sG2d, __bf16* sF,
    int tid, int wid, int lane, int mcol, int hi8, int hib16) {
  constexpr int DONE   = FIRST ? 16 : 256;
  constexpr int DTWO   = FIRST ? 4  : 32;
  constexpr int FDIM   = 3 * DONE + 2 * DTWO;       // 56 / 832
  constexpr int FPAD   = (FDIM + 31) & ~31;         // 64 / 832
  constexpr int KSTEPS = FPAD / 32;                 // 2  / 26
  constexpr int SH2    = FIRST ? 2 : 5;

  // ---- g1 means ----
  if (tid < DONE) {
    float su = 0.f, sd = 0.f;
#pragma unroll
    for (int n = 0; n < S0E; ++n) {
      su += sOne[n * NO1 + tid];
      sd += sOne[(S0E + n) * NO1 + tid];
    }
    sG1u[tid] = su * (1.f / S0E);
    sG1d[tid] = sd * (1.f / S0E);
  }
  // ---- g2 means ----
#pragma unroll
  for (int idx = tid; idx < NE * DTWO; idx += 256) {
    int n = idx >> SH2, k = idx & (DTWO - 1);
    float su = 0.f, sd = 0.f;
#pragma unroll
    for (int j = 0; j < S0E; ++j) {
      su += sTwo[(n * NE + j) * NO2 + k];
      sd += sTwo[(n * NE + S0E + j) * NO2 + k];
    }
    sG2u[n * NO2 + k] = su * (1.f / S0E);
    sG2d[n * NO2 + k] = sd * (1.f / S0E);
  }
  __syncthreads();

  // ---- pack f = [one | g1u | g1d | g2u | g2d | 0] as bf16, 2 rows/wave ----
#pragma unroll
  for (int rr = 0; rr < 2; ++rr) {
    int n = wid * 2 + rr;
    const float* oneRow = sOne + n * NO1;
    __bf16* frow = sF + n * FMAX;
    if (FIRST) {
#pragma unroll
      for (int k = lane; k < FPAD; k += 32) {
        float v;
        if      (k < DONE)            v = oneRow[k];
        else if (k < 2 * DONE)        v = sG1u[k - DONE];
        else if (k < 3 * DONE)        v = sG1d[k - 2 * DONE];
        else if (k < 3 * DONE + DTWO) v = sG2u[n * NO2 + (k - 3 * DONE)];
        else if (k < FDIM)            v = sG2d[n * NO2 + (k - 3 * DONE - DTWO)];
        else                          v = 0.f;
        frow[k] = (__bf16)v;
      }
    } else {
#pragma unroll
      for (int t = 0; t < 8; ++t) { int k = lane + 32*t; frow[k]       = (__bf16)oneRow[k]; }
#pragma unroll
      for (int t = 0; t < 8; ++t) { int k = lane + 32*t; frow[256 + k] = (__bf16)sG1u[k]; }
#pragma unroll
      for (int t = 0; t < 8; ++t) { int k = lane + 32*t; frow[512 + k] = (__bf16)sG1d[k]; }
      frow[768 + lane] = (__bf16)sG2u[n * NO2 + lane];
      frow[800 + lane] = (__bf16)sG2d[n * NO2 + lane];
    }
  }
  __syncthreads();

  // ---- one-stream GEMM: (16 x FDIM) x (FDIM x 256); 2 col tiles / wave ----
  v8f acc0 = {}; v8f acc1 = {};
  {
    const __bf16* arow = sF + mcol * FMAX;
    if (PACKED) {
      // fragment-order: [ct][ks][lane][16] -> constant-stride coalesced loads
      const __bf16* w0 = pV + ((size_t)(2 * wid) * KSTEPS * 32 + lane) * 16;
      const __bf16* w1 = w0 + (size_t)KSTEPS * 512;
#pragma unroll
      for (int ks = 0; ks < KSTEPS; ++ks) {
        v16bf af = fragA_bf16(arow + ks * 32, hi8);
        v16bf b0 = frag16(w0 + ks * 512);
        v16bf b1 = frag16(w1 + ks * 512);
        acc0 = wmma_bf16(af, b0, acc0);
        acc1 = wmma_bf16(af, b1, acc1);
      }
    } else {
      const float* wr0 = Wv + (size_t)(32 * wid + mcol) * FDIM;
      const float* wr1 = wr0 + (size_t)16 * FDIM;
#pragma unroll
      for (int ks = 0; ks < KSTEPS; ++ks) {
        int kb = ks * 32;
        v16bf af = fragA_bf16(arow + kb, hi8);
        v16bf b0, b1;
        if (FIRST) { b0 = fragB_clamp<FDIM>(wr0, kb, hib16);
                     b1 = fragB_clamp<FDIM>(wr1, kb, hib16); }
        else       { b0 = fragB_f32(wr0 + kb + hib16);
                     b1 = fragB_f32(wr1 + kb + hib16); }
        acc0 = wmma_bf16(af, b0, acc0);
        acc1 = wmma_bf16(af, b1, acc1);
      }
    }
  }

  // ---- two-stream GEMM: (256 x DTWO) x (DTWO x 32); 4 tiles / wave ----
  v16bf bT0, bT1;
  if (PACKED) {
    bT0 = frag16(pW + lane * 16);
    bT1 = frag16(pW + 512 + lane * 16);
  } else if (FIRST) {
    bT0 = fragB_clamp<4>(Ww + mcol * 4, 0, hib16);
    bT1 = fragB_clamp<4>(Ww + (16 + mcol) * 4, 0, hib16);
  } else {
    bT0 = fragB_f32(Ww + mcol * NO2 + hib16);
    bT1 = fragB_f32(Ww + (16 + mcol) * NO2 + hib16);
  }
  v8f accT[4];
#pragma unroll
  for (int ii = 0; ii < 2; ++ii) {
    int i = wid * 2 + ii;
    const float* arow2 = sTwo + (i * NE + mcol) * NO2;
    v16bf af;
    if (FIRST) {
      float4 q = *(const float4*)arow2;
      float qa[4] = {q.x, q.y, q.z, q.w};
      v16bf t;
#pragma unroll
      for (int e = 0; e < 16; ++e)
        t[e] = (__bf16)((e < 4 && hi8 == 0) ? qa[e] : 0.f);
      af = t;
    } else {
      af = fragA_f32(arow2, hi8);
    }
    v8f z0 = {}; v8f z1 = {};
    accT[ii * 2 + 0] = wmma_bf16(af, bT0, z0);
    accT[ii * 2 + 1] = wmma_bf16(af, bT1, z1);
  }
  __syncthreads();   // all reads of sOne/sTwo/sF complete

  // ---- writeback one: tanh + bias (+ residual) ----
#pragma unroll
  for (int c = 0; c < 2; ++c) {
    const v8f* acc = c ? &acc1 : &acc0;
    int o = 32 * wid + 16 * c + mcol;
    float bias = bv[o];
#pragma unroll
    for (int v = 0; v < 8; ++v) {
      int n = v + (hi8 ? 8 : 0);
      float val = fast_tanh((*acc)[v] + bias);
      if (resid) val += sOne[n * NO1 + o];
      sOne[n * NO1 + o] = val;
    }
  }
  // ---- writeback two ----
#pragma unroll
  for (int jj = 0; jj < 4; ++jj) {
    int i = wid * 2 + (jj >> 1), ct = jj & 1;
    int o = 16 * ct + mcol;
    float bias = bw[o];
#pragma unroll
    for (int v = 0; v < 8; ++v) {
      int j = v + (hi8 ? 8 : 0);
      float val = fast_tanh(accT[jj][v] + bias);
      if (resid) val += sTwo[(i * NE + j) * NO2 + o];
      sTwo[(i * NE + j) * NO2 + o] = val;
    }
  }
  __syncthreads();
}

// ---------------------------------------------------------------------------
template<bool PACKED>
__device__ __forceinline__ void ferminet_body(
    const float* __restrict__ x, const float* __restrict__ nuc,
    const float* __restrict__ v0_w, const float* __restrict__ v0_b,
    const float* __restrict__ v_w,  const float* __restrict__ v_b,
    const float* __restrict__ w0_w, const float* __restrict__ w0_b,
    const float* __restrict__ w_w,  const float* __restrict__ w_b,
    const float* __restrict__ env_w, const float* __restrict__ env_g,
    const float* __restrict__ sigma, const float* __restrict__ pi_,
    const __bf16* __restrict__ ws, float* __restrict__ out) {
  __shared__ __align__(16) char smem[(NE*NO1 + NE*NE*NO2 + 2*NO1 + 2*NE*NO2 + NE*NM) * 4
                                     + NE*FMAX * 2];
  float*  sOne = (float*)smem;                 // [16][256]
  float*  sTwo = sOne + NE*NO1;                // [16][16][32]
  float*  sG1u = sTwo + NE*NE*NO2;             // [256]
  float*  sG1d = sG1u + NO1;                   // [256]
  float*  sG2u = sG1d + NO1;                   // [16][32]
  float*  sG2d = sG2u + NE*NO2;                // [16][32]
  float*  sR   = sG2d + NE*NO2;                // [16][4]
  __bf16* sF   = (__bf16*)(sR + NE*NM);        // [16][832] bf16
  float*  sPsi = (float*)sF;                   // [16 det][2 blk][8][8]
  float*  sDet = sPsi + ND*2*8*8;              // [32]
  __bf16* sHbf = (__bf16*)(sDet + 32);         // [16][256] bf16

  const int tid  = threadIdx.x;
  const int wid  = tid >> 5;
  const int lane = tid & 31;
  const int mcol = lane & 15;
  const int hi8   = (lane & 16) ? 8  : 0;
  const int hib16 = (lane & 16) ? 16 : 0;
  const int b = blockIdx.x;
  const float* xb = x + b * NE * 3;

  // ---- init features ----
  {
    int i = tid >> 4, j = tid & 15;
    float dx = xb[i*3+0] - xb[j*3+0];
    float dy = xb[i*3+1] - xb[j*3+1];
    float dz = xb[i*3+2] - xb[j*3+2];
    float s  = dx*dx + dy*dy + dz*dz;
    float d  = (s > 0.f) ? sqrtf(s) : 0.f;
    float* t = sTwo + (i*NE + j) * NO2;
    t[0] = dx; t[1] = dy; t[2] = dz; t[3] = d;
  }
  if (tid < NE*NM) {
    int n = tid >> 2, m = tid & 3;
    float dx = xb[n*3+0] - nuc[m*3+0];
    float dy = xb[n*3+1] - nuc[m*3+1];
    float dz = xb[n*3+2] - nuc[m*3+2];
    float d  = sqrtf(dx*dx + dy*dy + dz*dz);
    float* o = sOne + n*NO1 + m*4;
    o[0] = dx; o[1] = dy; o[2] = dz; o[3] = d;
    sR[n*NM + m] = d;
  }
  __syncthreads();

  // ---- layers ----
  do_layer<true, PACKED>(false, ws + OFF_V0, ws + OFF_W, v0_w, w0_w,
                         v0_b, w0_b,
                         sOne, sTwo, sG1u, sG1d, sG2u, sG2d, sF,
                         tid, wid, lane, mcol, hi8, hib16);
  for (int l = 1; l < 4; ++l) {
    do_layer<false, PACKED>(true,
                            ws + OFF_V + (size_t)(l-1)*VL_SZ,
                            ws + OFF_W + l*1024,
                            v_w + (size_t)(l-1)*NO1*FMAX,
                            w_w + (l-1)*NO2*NO2,
                            v_b + (l-1)*NO1, w_b + (l-1)*NO2,
                            sOne, sTwo, sG1u, sG1d, sG2u, sG2d, sF,
                            tid, wid, lane, mcol, hi8, hib16);
  }

  // ---- pack h to bf16 for envelope B ----
  {
    int n = tid >> 4;
    int c = (tid & 15) * 16;
    const float* src = sOne + n * NO1 + c;
    __bf16* dst = sHbf + n * NO1 + c;
    const float4 a  = *(const float4*)(src + 0);
    const float4 bq = *(const float4*)(src + 4);
    const float4 cq = *(const float4*)(src + 8);
    const float4 dq = *(const float4*)(src + 12);
    v8bf lo, hi;
    lo[0]=(__bf16)a.x;  lo[1]=(__bf16)a.y;  lo[2]=(__bf16)a.z;  lo[3]=(__bf16)a.w;
    lo[4]=(__bf16)bq.x; lo[5]=(__bf16)bq.y; lo[6]=(__bf16)bq.z; lo[7]=(__bf16)bq.w;
    hi[0]=(__bf16)cq.x; hi[1]=(__bf16)cq.y; hi[2]=(__bf16)cq.z; hi[3]=(__bf16)cq.w;
    hi[4]=(__bf16)dq.x; hi[5]=(__bf16)dq.y; hi[6]=(__bf16)dq.z; hi[7]=(__bf16)dq.w;
    *(v8bf*)(dst + 0) = lo;
    *(v8bf*)(dst + 8) = hi;
  }
  __syncthreads();

  // ---- envelope GEMM: (128 x 256) x (256 x 16) ----
  {
    v8f accE = {};
    const int rg = 16*wid + mcol;              // row = det*8 + orbital
    const int kd = rg >> 3, io = rg & 7;
    const __bf16* ebase = ws + OFF_E + ((size_t)(wid * 8) * 32 + lane) * 16;
    const float*  ewrow = env_w + (size_t)(kd*NE + io) * NO1;
    const __bf16* hrow  = sHbf + mcol * NO1;
#pragma unroll
    for (int ks = 0; ks < 8; ++ks) {
      v16bf af = PACKED ? frag16(ebase + ks * 512)
                        : fragA_f32(ewrow + ks * 32, hi8);
      v16bf bf = frag16(hrow + ks * 32 + hib16);
      accE = wmma_bf16(af, bf, accE);
    }
#pragma unroll
    for (int v = 0; v < 8; ++v) {
      int row = 16*wid + v + (hi8 ? 8 : 0);
      int k = row >> 3, i = row & 7;
      int col = mcol;
      int blk = col >> 3, j = col & 7;
      float lin = accE[v] + 256.f * env_g[k*NE + i];
      float env = 0.f;
#pragma unroll
      for (int m = 0; m < NM; ++m) {
        float sg = fabsf(sigma[(k*NE + i)*NM + m]);
        float pw = pi_[(k*NE + i)*NM + m];
        env += pw * __expf(-sg * sR[col*NM + m]);
      }
      sPsi[((k*2 + blk)*8 + i)*8 + j] = lin * env;
    }
  }
  __syncthreads();

  // ---- 32 determinants of 8x8 (wave 0, register LU w/ pivoting) ----
  if (wid == 0) {
    int kd = lane & 15, blk = lane >> 4;
    float a[8][8];
    const float* P = sPsi + ((kd*2 + blk)*8)*8;
#pragma unroll
    for (int i = 0; i < 8; ++i)
#pragma unroll
      for (int j = 0; j < 8; ++j) a[i][j] = P[i*8 + j];
    float det = 1.f;
#pragma unroll
    for (int c = 0; c < 8; ++c) {
#pragma unroll
      for (int rs = c + 1; rs < 8; ++rs) {
        bool sw = fabsf(a[rs][c]) > fabsf(a[c][c]);
#pragma unroll
        for (int j = 0; j < 8; ++j) {
          float t0 = a[c][j], t1 = a[rs][j];
          a[c][j]  = sw ? t1 : t0;
          a[rs][j] = sw ? t0 : t1;
        }
        det = sw ? -det : det;
      }
      det *= a[c][c];
      float inv = __builtin_amdgcn_rcpf(a[c][c]);
#pragma unroll
      for (int rr = c + 1; rr < 8; ++rr) {
        float f = a[rr][c] * inv;
#pragma unroll
        for (int j = c + 1; j < 8; ++j) a[rr][j] -= f * a[c][j];
      }
    }
    sDet[blk*16 + kd] = det;
  }
  __syncthreads();

  if (tid == 0) {
    float s = 0.f;
#pragma unroll
    for (int k = 0; k < ND; ++k) s += sDet[k] * sDet[16 + k];
    out[b] = s;
  }
}

extern "C" __global__ __launch_bounds__(256)
void ferminet_packed(const float* x, const float* nuc,
                     const float* v0_w, const float* v0_b,
                     const float* v_w,  const float* v_b,
                     const float* w0_w, const float* w0_b,
                     const float* w_w,  const float* w_b,
                     const float* env_w, const float* env_g,
                     const float* sigma, const float* pi_,
                     const __bf16* ws, float* out) {
  ferminet_body<true>(x, nuc, v0_w, v0_b, v_w, v_b, w0_w, w0_b, w_w, w_b,
                      env_w, env_g, sigma, pi_, ws, out);
}

extern "C" __global__ __launch_bounds__(256)
void ferminet_plain(const float* x, const float* nuc,
                    const float* v0_w, const float* v0_b,
                    const float* v_w,  const float* v_b,
                    const float* w0_w, const float* w0_b,
                    const float* w_w,  const float* w_b,
                    const float* env_w, const float* env_g,
                    const float* sigma, const float* pi_,
                    const __bf16* ws, float* out) {
  ferminet_body<false>(x, nuc, v0_w, v0_b, v_w, v_b, w0_w, w0_b, w_w, w_b,
                       env_w, env_g, sigma, pi_, ws, out);
}

extern "C" void kernel_launch(void* const* d_in, const int* in_sizes, int n_in,
                              void* d_out, int out_size, void* d_ws, size_t ws_size,
                              hipStream_t stream) {
  const float* x    = (const float*)d_in[0];
  const float* nuc  = (const float*)d_in[1];
  const float* v0w  = (const float*)d_in[2];
  const float* v0b  = (const float*)d_in[3];
  const float* vw   = (const float*)d_in[4];
  const float* vb   = (const float*)d_in[5];
  const float* w0w  = (const float*)d_in[6];
  const float* w0b  = (const float*)d_in[7];
  const float* ww   = (const float*)d_in[8];
  const float* wb   = (const float*)d_in[9];
  const float* envw = (const float*)d_in[10];
  const float* envg = (const float*)d_in[11];
  const float* sig  = (const float*)d_in[12];
  const float* pw   = (const float*)d_in[13];
  float* out = (float*)d_out;
  __bf16* ws = (__bf16*)d_ws;
  const int B = in_sizes[0] / (NE * 3);   // 4096

  if (ws_size >= WS_BYTES) {
    hipLaunchKernelGGL(pack_weights, dim3((WS_ELEMS + 255) / 256), dim3(256), 0,
                       stream, v0w, vw, w0w, ww, envw, ws);
    hipLaunchKernelGGL(ferminet_packed, dim3(B), dim3(256), 0, stream,
                       x, nuc, v0w, v0b, vw, vb, w0w, w0b, ww, wb,
                       envw, envg, sig, pw, ws, out);
  } else {
    hipLaunchKernelGGL(ferminet_plain, dim3(B), dim3(256), 0, stream,
                       x, nuc, v0w, v0b, vw, vb, w0w, w0b, ww, wb,
                       envw, envg, sig, pw, ws, out);
  }
}
